// CrossBaby_1_953482739798
// MI455X (gfx1250) — matrix operand order repaired and checked by reading, not compiled, not run on hardware
//
#include <hip/hip_runtime.h>
#include <hip/hip_bf16.h>

// ---------------------------------------------------------------------------
// CDNA5 / gfx1250 pipeline for the CrossBaby reference.  B=32,S=128,V=8192,E=256
// Heavy GEMMs: v_wmma_f32_16x16x32_bf16 (wave32).
// Stage 1 stages the x tile in LDS with global_load_async_to_lds_b128
// (ASYNCcnt), double buffered; B fragments are prefetched per K-chunk before
// the barrier so WMMAs don't stall on s_wait_loadcnt 0.
// ---------------------------------------------------------------------------

typedef __attribute__((ext_vector_type(16))) __bf16 v16bf;
typedef __attribute__((ext_vector_type(8)))  __bf16 v8bf;
typedef __attribute__((ext_vector_type(8)))  float  v8f;
typedef __attribute__((ext_vector_type(4)))  float  v4f;

static constexpr int Bn = 32, Sn = 128, Vn = 8192, En = 256;

// ---------------------------------------------------------------------------
// fp32 -> bf16 bulk convert (vectorized x8)
// ---------------------------------------------------------------------------
__global__ __launch_bounds__(256) void cvt_f32_bf16(const float* __restrict__ in,
                                                    __bf16* __restrict__ out, int n8) {
  int i = blockIdx.x * blockDim.x + threadIdx.x;
  if (i >= n8) return;
  v8f v = ((const v8f*)in)[i];
  v8bf o;
#pragma unroll
  for (int j = 0; j < 8; ++j) o[j] = (__bf16)v[j];
  ((v8bf*)out)[i] = o;
}

// ---------------------------------------------------------------------------
// Stage 1: h = relu(x @ w_emb^T + b_emb)   M=4096 K=8192 N=256
// Block = 256 thr = 8 waves; block tile 32(M) x 256(N); wave tile 16x64.
// A (fp32) staged in LDS via async copies (double buffered, KSTEP=64).
// Row stride padded to 68 floats -> ds_load_b128 banks (4*l mod 64).
// Per chunk: prefetch 8 B fragments -> issue next async copy ->
// s_wait_asynccnt 2 -> barrier -> LDS reads + cvt + 8 WMMAs -> barrier.
// ---------------------------------------------------------------------------
__global__ __launch_bounds__(256) void gemm_s1_async(
    const float* __restrict__ A, const __bf16* __restrict__ B,
    const float* __restrict__ bias, __bf16* __restrict__ C,
    int K, int lda, int ldb, int ldc) {
  constexpr int KSTEP = 64;
  constexpr int ROWP = 68;                    // padded row stride (floats)
  __shared__ float tile[2][32 * ROWP];        // 2 x 8704 B

  const int t = threadIdx.x;
  const int wave = t >> 5, lane = t & 31;
  const int lo = lane & 15, hi = lane >> 4;
  const int rowL0 = (wave >> 2) * 16;         // wave's M offset inside block tile
  const int col0 = (wave & 3) * 64;           // N strip (N == 256, gridDim.x == 1)

  // async-copy mapping: thread moves 32B of the 32x64 fp32 tile per chunk
  const int crow = t >> 3;                    // 0..31
  const int ccol = (t & 7) * 8;               // 0,8,..,56 (floats)
  const float* saddrA = A + (size_t)(blockIdx.y * 32) * lda;        // uniform SGPR pair
  const unsigned voffBase = (unsigned)(crow * lda + ccol) * 4u;     // per-lane byte off
  const unsigned ldsOff = (unsigned)(crow * ROWP + ccol) * 4u;
  const unsigned lds0 = ((unsigned)(uintptr_t)&tile[0][0]) + ldsOff;
  const unsigned lds1 = ((unsigned)(uintptr_t)&tile[1][0]) + ldsOff;

  auto issue = [&](int kk, int bufsel) {
    const unsigned voff = voffBase + (unsigned)kk * 4u;
    const unsigned lds = bufsel ? lds1 : lds0;
    // INST_OFFSET is added to BOTH the LDS and global address (ISA 10.7.x),
    // so offset:16 fetches the second b128 of this thread's 32B strip.
    asm volatile("global_load_async_to_lds_b128 %0, %1, %2"
                 :: "v"(lds), "v"(voff), "s"(saddrA) : "memory");
    asm volatile("global_load_async_to_lds_b128 %0, %1, %2 offset:16"
                 :: "v"(lds), "v"(voff), "s"(saddrA) : "memory");
  };

  // Per-u B base pointers (lane holds column col0+u*16+lo, 16 contiguous K)
  const __bf16* bp[4];
#pragma unroll
  for (int u = 0; u < 4; ++u)
    bp[u] = B + (size_t)(col0 + u * 16 + lo) * ldb + 16 * hi;

  v8f acc[4] = {};
  const int nchunks = K / KSTEP;
  issue(0, 0);
  for (int c = 0; c < nchunks; ++c) {
    const int kk = c * KSTEP;

    // ---- prefetch all 8 B fragments of this chunk (global; no LDS dep)
    v16bf bfr[8];
#pragma unroll
    for (int ks = 0; ks < 2; ++ks)
#pragma unroll
      for (int u = 0; u < 4; ++u)
        bfr[ks * 4 + u] = *(const v16bf*)(bp[u] + kk + ks * 32);

    if (c + 1 < nchunks) {
      issue(kk + KSTEP, (c + 1) & 1);                 // overlap next copy
      asm volatile("s_wait_asynccnt 2" ::: "memory"); // chunk c resident
    } else {
      asm volatile("s_wait_asynccnt 0" ::: "memory");
    }
    __syncthreads();

    const float* tb = &tile[c & 1][0];
#pragma unroll
    for (int ks = 0; ks < 2; ++ks) {
      // A fragment (16x32): lane(lo,hi) row rowL0+lo,
      // K = 32*ks + 16*hi + [0..15]
      const float* lp = tb + ((rowL0 + lo) * ROWP + ks * 32 + 16 * hi);
      v4f a0 = *(const v4f*)lp;
      v4f a1 = *(const v4f*)(lp + 4);
      v4f a2 = *(const v4f*)(lp + 8);
      v4f a3 = *(const v4f*)(lp + 12);
      v16bf af;
#pragma unroll
      for (int i = 0; i < 4; ++i) {
        af[i]      = (__bf16)a0[i];
        af[4 + i]  = (__bf16)a1[i];
        af[8 + i]  = (__bf16)a2[i];
        af[12 + i] = (__bf16)a3[i];
      }
#pragma unroll
      for (int u = 0; u < 4; ++u)
        acc[u] = __builtin_amdgcn_wmma_f32_16x16x32_bf16(
            false, af, false, bfr[ks * 4 + u], (short)0, acc[u], false, false);
    }
    __syncthreads();   // buffer (c&1) free before overwrite at iter c+1
  }

  // epilogue: C VGPR v -> (M = row0 + v + 8*hi, N = col0 + u*16 + lo)
  const int row0 = blockIdx.y * 32 + rowL0;
#pragma unroll
  for (int u = 0; u < 4; ++u) {
    const int cc = col0 + u * 16 + lo;
    const float bs = bias[cc];
#pragma unroll
    for (int v = 0; v < 8; ++v) {
      float r = fmaxf(acc[u][v] + bs, 0.0f);
      C[(size_t)(row0 + v + 8 * hi) * ldc + cc] = (__bf16)r;
    }
  }
}

// ---------------------------------------------------------------------------
// Generic K-major bf16 WMMA GEMM (stages 4/5/6; operands small / L2-resident):
//   C[M,N] = act( A[M,K] * B[N,K]^T + bias[N] )
// ---------------------------------------------------------------------------
template <bool RELU, typename OT>
__global__ __launch_bounds__(256) void gemm_wmma(
    const __bf16* __restrict__ A, const __bf16* __restrict__ B,
    const float* __restrict__ bias, OT* __restrict__ C,
    int M, int N, int K, int lda, int ldb, int ldc,
    long strideA, long strideB, long strideC) {
  const int z = blockIdx.z;
  A += (long)z * strideA;
  B += (long)z * strideB;
  C += (long)z * strideC;

  const int wave = threadIdx.x >> 5;
  const int lane = threadIdx.x & 31;
  const int lo = lane & 15, hi = lane >> 4;

  const int row0 = blockIdx.y * 32 + (wave >> 2) * 16;
  const int col0 = blockIdx.x * 256 + (wave & 3) * 64;
  if (row0 >= M || col0 >= N) return;   // wave-uniform

  const __bf16* ap = A + (size_t)(row0 + lo) * lda + 16 * hi;
  const __bf16* bp[4];
#pragma unroll
  for (int u = 0; u < 4; ++u)
    bp[u] = B + (size_t)(col0 + u * 16 + lo) * ldb + 16 * hi;

  v8f acc[4] = {};
#pragma unroll 2
  for (int kk = 0; kk < K; kk += 32) {
    // hoist the 4 B fragments + A chunks: one clause, staggered waits
    v16bf bfr[4];
#pragma unroll
    for (int u = 0; u < 4; ++u) bfr[u] = *(const v16bf*)(bp[u] + kk);
    v8bf a0 = *(const v8bf*)(ap + kk);
    v8bf a1 = *(const v8bf*)(ap + kk + 8);
    v16bf af;
#pragma unroll
    for (int i = 0; i < 8; ++i) { af[i] = a0[i]; af[8 + i] = a1[i]; }
#pragma unroll
    for (int u = 0; u < 4; ++u)
      acc[u] = __builtin_amdgcn_wmma_f32_16x16x32_bf16(
          false, af, false, bfr[u], (short)0, acc[u], false, false);
  }

#pragma unroll
  for (int u = 0; u < 4; ++u) {
    const int cc = col0 + u * 16 + lo;
    const float bs = bias ? bias[cc] : 0.0f;
#pragma unroll
    for (int v = 0; v < 8; ++v) {
      float r = acc[u][v] + bs;
      if (RELU) r = fmaxf(r, 0.0f);
      C[(size_t)(row0 + v + 8 * hi) * ldc + cc] = (OT)r;
    }
  }
}

// ---------------------------------------------------------------------------
// Stage 2: s[b,i] = sum_j h[b,i,j]
// ---------------------------------------------------------------------------
__global__ __launch_bounds__(256) void rowsum256(const __bf16* __restrict__ h,
                                                 float* __restrict__ s, int rows) {
  int r = blockIdx.x * blockDim.x + threadIdx.x;
  if (r >= rows) return;
  const v8bf* p = (const v8bf*)(h + (size_t)r * En);
  float acc = 0.0f;
#pragma unroll 4
  for (int i = 0; i < En / 8; ++i) {
    v8bf v = p[i];
#pragma unroll
    for (int j = 0; j < 8; ++j) acc += (float)v[j];
  }
  s[r] = acc;
}

// ---------------------------------------------------------------------------
// Stage 3: Weff[b, e*256+j] = sum_k w_red[e, k*256+j] * s[b,k]   (bf16 out)
// block = e, thread = j; coalesced fp32 reads of w_red reused across all 32
// batches via 32 per-thread accumulators; s broadcast from LDS.
// ---------------------------------------------------------------------------
__global__ __launch_bounds__(256) void weff_kernel(const float* __restrict__ w_red,
                                                   const float* __restrict__ s,
                                                   __bf16* __restrict__ weff) {
  __shared__ float sl[Bn * Sn];           // 16 KiB
  const int e = blockIdx.x, j = threadIdx.x;
  for (int i = threadIdx.x; i < Bn * Sn; i += blockDim.x) sl[i] = s[i];
  __syncthreads();

  float acc[Bn] = {};
  const float* wp = w_red + (size_t)e * (Sn * En) + j;
  for (int k = 0; k < Sn; ++k) {
    const float w = wp[(size_t)k * En];
#pragma unroll
    for (int b = 0; b < Bn; ++b) acc[b] += sl[b * Sn + k] * w;
  }
#pragma unroll
  for (int b = 0; b < Bn; ++b)
    weff[(size_t)b * (En * En) + e * En + j] = (__bf16)acc[b];
}

// ---------------------------------------------------------------------------
// Stage 5 reduce: y2[i] = relu(sum_c part[c][i] + b_red2[i%256]) -> bf16
// ---------------------------------------------------------------------------
__global__ __launch_bounds__(256) void reduce_splitk(const float* __restrict__ part,
                                                     const float* __restrict__ bias,
                                                     __bf16* __restrict__ y2, int chunks) {
  int i = blockIdx.x * blockDim.x + threadIdx.x;   // 0 .. B*E-1
  float a = 0.0f;
  for (int c = 0; c < chunks; ++c) a += part[(size_t)c * (Bn * En) + i];
  a += bias[i & (En - 1)];
  y2[i] = (__bf16)fmaxf(a, 0.0f);
}

// ---------------------------------------------------------------------------
// Launch
// ---------------------------------------------------------------------------
extern "C" void kernel_launch(void* const* d_in, const int* in_sizes, int n_in,
                              void* d_out, int out_size, void* d_ws, size_t ws_size,
                              hipStream_t stream) {
  const float* x      = (const float*)d_in[0];   // [B,S,V]
  const float* w_emb  = (const float*)d_in[1];   // [E,V]
  const float* b_emb  = (const float*)d_in[2];   // [E]
  const float* w_red  = (const float*)d_in[3];   // [E, S*E]
  const float* b_red  = (const float*)d_in[4];   // [E]
  const float* w_red2 = (const float*)d_in[5];   // [E, S*E]
  const float* b_red2 = (const float*)d_in[6];   // [E]
  const float* w_out  = (const float*)d_in[7];   // [V,E]
  const float* b_out  = (const float*)d_in[8];   // [V]
  float* out = (float*)d_out;                    // [B,V]

  // Workspace layout (bytes), 256B aligned, ~33 MiB total
  char* ws = (char*)d_ws;
  __bf16* web  = (__bf16*)(ws + 0);              //  4 MiB  w_emb bf16 [E,V]
  __bf16* h    = (__bf16*)(ws + 4194304);        //  2 MiB  [B*S, E]
  float*  sfp  = (float*) (ws + 6291456);        // 16 KiB  [B*S]
  __bf16* weff = (__bf16*)(ws + 6307840);        //  4 MiB  [B, E*E]
  __bf16* y    = (__bf16*)(ws + 10502144);       //  2 MiB  [B*S, E]
  __bf16* wr2b = (__bf16*)(ws + 12599296);       // 16 MiB  w_red2 bf16 [E, S*E]
  __bf16* y2   = (__bf16*)(ws + 29376512);       // 16 KiB  [B, E]
  __bf16* wob  = (__bf16*)(ws + 29392896);       //  4 MiB  w_out bf16 [V,E]
  float*  part = (float*) (ws + 33587200);       //  1 MiB  split-K partials

  const dim3 blk(256);

  // Weight conversions (fp32 -> bf16)
  cvt_f32_bf16<<<dim3((En * Vn / 8 + 255) / 256), blk, 0, stream>>>(w_emb, web, En * Vn / 8);
  cvt_f32_bf16<<<dim3((En * Sn * En / 8 + 255) / 256), blk, 0, stream>>>(w_red2, wr2b, En * Sn * En / 8);
  cvt_f32_bf16<<<dim3((Vn * En / 8 + 255) / 256), blk, 0, stream>>>(w_out, wob, Vn * En / 8);

  // Stage 1: h = relu(x @ w_emb^T + b_emb), async-LDS staged A
  gemm_s1_async<<<dim3(1, (Bn * Sn) / 32, 1), blk, 0, stream>>>(
      x, web, b_emb, h, Vn, Vn, Vn, En);

  // Stage 2: s = h.sum(-1)
  rowsum256<<<dim3(Bn * Sn / 256), blk, 0, stream>>>(h, sfp, Bn * Sn);

  // Stage 3: Weff[b,e,j] = sum_k w_red[e,k,j] * s[b,k]
  weff_kernel<<<dim3(En), blk, 0, stream>>>(w_red, sfp, weff);

  // Stage 4: y_b = relu(h_b @ Weff_b^T + b_red), batched over b
  gemm_wmma<true, __bf16><<<dim3(1, Sn / 32, Bn), blk, 0, stream>>>(
      h, weff, b_red, y, Sn, En, En, En, En, En,
      (long)Sn * En, (long)En * En, (long)Sn * En);

  // Stage 5: split-K x32 (M=32 K=32768 N=256; chunk K=1024 via batch strides)
  gemm_wmma<false, float><<<dim3(1, 1, 32), blk, 0, stream>>>(
      y, wr2b, nullptr, part, Bn, En, Sn * En / 32, Sn * En, Sn * En, En,
      1024L, 1024L, (long)Bn * En);
  reduce_splitk<<<dim3(Bn * En / 256), blk, 0, stream>>>(part, b_red2, y2, 32);

  // Stage 6: out = y2 @ w_out^T + b_out   M=32 N=8192 K=256
  gemm_wmma<false, float><<<dim3(Vn / 256, 1, 1), blk, 0, stream>>>(
      y2, wob, b_out, out, Bn, Vn, En, En, En, Vn, 0, 0, 0);
}
